// LSTM_Layer_89790586290707
// MI455X (gfx1250) — compile-verified, matrix operand order checked by reading, not run on hardware
//
#include <hip/hip_runtime.h>
#include <hip/hip_bf16.h>
#include <math.h>

// LSTM layer for MI455X (gfx1250, wave32, WMMA).
//  B=16, T=2048, I=H=512, G=4H=2048.
//
// Phase 1: wx[b,t,:] = x[b,t,:] @ W^T + w_bias  (bulk GEMM, bf16 WMMA, f32 acc,
//          4Mx4N register blocking -> ~16 FLOP per L2 byte)
// Phase 2: persistent 16-WG kernel; U kept in VGPRs as WMMA B-fragments
//          (128 waves x 16 v16bf frags = entire 2MB bf16 U in registers),
//          one device-wide atomic barrier per timestep, h exchanged via a
//          double-buffered global bf16 buffer staged into LDS with
//          GLOBAL_LOAD_ASYNC_TO_LDS_B128 (ASYNCcnt-tracked, VGPR-bypassing).

typedef __attribute__((ext_vector_type(16))) __bf16 v16bf;
typedef __attribute__((ext_vector_type(8)))  float  v8f;
typedef __attribute__((ext_vector_type(4)))  int    v4i;

#define B_ 16
#define T_ 2048
#define I_ 512
#define H_ 512
#define G_ 2048  // 4*H

#if defined(__HIP_DEVICE_COMPILE__) &&                                   \
    __has_builtin(__builtin_amdgcn_global_load_async_to_lds_b128) &&     \
    __has_builtin(__builtin_amdgcn_s_wait_asynccnt)
#define USE_ASYNC_LDS 1
typedef __attribute__((address_space(1))) v4i gv4i;  // global int4
typedef __attribute__((address_space(3))) v4i lv4i;  // LDS int4
#else
#define USE_ASYNC_LDS 0
#endif

// Build a 16x32 bf16 WMMA operand fragment from two contiguous 8-float runs.
// CDNA5 16-bit A-matrix layout: lane<16 holds K = kb..kb+7 and kb+16..kb+23
// (kb = kt*32); lanes>=16 the +8-shifted halves (caller folds via lhi*8).
__device__ inline v16bf frag_from_f32(const float* __restrict__ p0,
                                      const float* __restrict__ p1) {
  v16bf r;
#pragma unroll
  for (int e = 0; e < 8; ++e) {
    r[e]     = (__bf16)p0[e];
    r[e + 8] = (__bf16)p1[e];
  }
  return r;
}

// Same fragment sourced from LDS-resident bf16: two ds_load_b128.
__device__ inline v16bf frag_from_lds(const __bf16* p0, const __bf16* p1) {
  union { v16bf v; ulonglong2 q[2]; } u;
  u.q[0] = *(const ulonglong2*)(const void*)p0;
  u.q[1] = *(const ulonglong2*)(const void*)p1;
  return u.v;
}

// ---------------------------------------------------------------------------
// Phase 0: init h double-buffer (slot 0) from h0 as bf16.
// ---------------------------------------------------------------------------
__global__ void init_kernel(const float* __restrict__ h0,
                            __bf16* __restrict__ hbuf) {
  for (int i = threadIdx.x; i < B_ * H_; i += blockDim.x)
    hbuf[i] = (__bf16)h0[i];
}

// ---------------------------------------------------------------------------
// Phase 1: wx = x @ W^T + w_bias.  32768x2048 output, K=512.
// Each wave computes a 64x64 block (4 M-tiles x 4 N-tiles, 16 f32 accums).
// 2048 blocks x 256 threads (8 waves/block).
// ---------------------------------------------------------------------------
__global__ __launch_bounds__(256) void gemm_wx_kernel(
    const float* __restrict__ x,   // [32768][512]   rows = b*T + t
    const float* __restrict__ w,   // [2048][512]
    const float* __restrict__ wb,  // [2048]
    float* __restrict__ wx)        // [32768][2048]
{
  const int wid  = blockIdx.x * 8 + (threadIdx.x >> 5);
  const int lane = threadIdx.x & 31;
  const int lrow = lane & 15;
  const int lhi  = lane >> 4;
  const int mb = wid & 511;  // 512 blocks of 64 rows
  const int nb = wid >> 9;   // 32 blocks of 64 cols
  const int m0 = mb * 64;
  const int n0 = nb * 64;

  v8f acc[4][4] = {};

  const float* xrow[4];
#pragma unroll
  for (int i = 0; i < 4; ++i) xrow[i] = x + (size_t)(m0 + i * 16 + lrow) * I_;
  const float* wrow[4];
#pragma unroll
  for (int j = 0; j < 4; ++j) wrow[j] = w + (size_t)(n0 + j * 16 + lrow) * I_;

#pragma unroll 2
  for (int kt = 0; kt < 16; ++kt) {
    const int kb = kt * 32 + lhi * 8;
    v16bf a[4];
#pragma unroll
    for (int i = 0; i < 4; ++i)
      a[i] = frag_from_f32(xrow[i] + kb, xrow[i] + kb + 16);
#pragma unroll
    for (int j = 0; j < 4; ++j) {
      const v16bf b = frag_from_f32(wrow[j] + kb, wrow[j] + kb + 16);
#pragma unroll
      for (int i = 0; i < 4; ++i)
        acc[i][j] = __builtin_amdgcn_wmma_f32_16x16x32_bf16(
            false, a[i], false, b, (short)0, acc[i][j], false, false);
    }
  }

  // C layout: VGPR r -> M = r + 8*lhi, lane&15 -> N.
#pragma unroll
  for (int j = 0; j < 4; ++j) {
    const int n = n0 + j * 16 + lrow;
    const float bias = wb[n];
#pragma unroll
    for (int i = 0; i < 4; ++i)
#pragma unroll
      for (int r = 0; r < 8; ++r) {
        const int row = m0 + i * 16 + lhi * 8 + r;
        wx[(size_t)row * G_ + n] = acc[i][j][r] + bias;
      }
  }
}

// ---------------------------------------------------------------------------
// Phase 2: persistent recurrent scan.
// 16 WGs x 256 threads (8 waves). WG wg owns h columns [wg*32, wg*32+32).
// Wave v: gate = v>>1 (i,f,g,o), subtile = v&1 -> gate-row base
//   n0 = gate*512 + wg*32 + subtile*16.
// U B-fragments for the wave's 16 columns live in 128 VGPRs for all T steps.
// One atomic grid barrier per step; h exchanged via global bf16 double buffer.
// ---------------------------------------------------------------------------
__global__ __launch_bounds__(256) void lstm_scan_kernel(
    const float* __restrict__ wx,    // [b*T+t][G]
    const float* __restrict__ u,     // [2048][512]
    const float* __restrict__ ubv,   // [2048]
    const float* __restrict__ c0,    // [16][512]
    __bf16* __restrict__ hbuf,       // [2][16*512] double buffer
    unsigned* __restrict__ bar_cnt,  // zeroed by host each launch
    float* __restrict__ out)         // h [B][T][H] then c [B][T][H]
{
  __shared__ __align__(16) __bf16 h_lds[B_ * H_];   // 16 KB
  __shared__ float gates_lds[8][256];               // 8 KB: [wave][M*16+N]

  const int tid  = threadIdx.x;
  const int wv   = tid >> 5;
  const int lane = tid & 31;
  const int lrow = lane & 15;
  const int lhi  = lane >> 4;
  const int wg   = blockIdx.x;  // 0..15
  const int hc0  = wg * 32;
  const int gate = wv >> 1;
  const int tile = wv & 1;
  const int nme  = gate * H_ + hc0 + tile * 16 + lrow;  // my gate row

  // --- preload U B-fragments (bf16) into registers: 16 k-tiles ---
  v16bf ufrag[16];
  {
    const float* urow = u + (size_t)nme * H_;
#pragma unroll
    for (int kt = 0; kt < 16; ++kt) {
      const int kb = kt * 32 + lhi * 8;
      ufrag[kt] = frag_from_f32(urow + kb, urow + kb + 16);
    }
  }
  const float ubias = ubv[nme];

  // --- per-thread cell state: 2 elements of c ---
  const int e0   = tid * 2;
  const int b_el = e0 >> 5;   // batch index
  const int cc   = e0 & 31;   // column within the WG's 32-col slice
  float c_reg[2];
  c_reg[0] = c0[b_el * H_ + hc0 + cc];
  c_reg[1] = c0[b_el * H_ + hc0 + cc + 1];

  float* out_h = out;
  float* out_c = out + (size_t)B_ * T_ * H_;

  for (int t = 0; t < T_; ++t) {
    // ---- stage h(t) (16 KB bf16) from global double-buffer into LDS ----
#if USE_ASYNC_LDS
    {
      gv4i* gsrc = (gv4i*)(void*)(hbuf + (size_t)(t & 1) * (B_ * H_)) + tid * 4;
      lv4i* ldst = (lv4i*)(void*)h_lds + tid * 4;
#pragma unroll
      for (int q = 0; q < 4; ++q)
        __builtin_amdgcn_global_load_async_to_lds_b128(gsrc + q, ldst + q, 0, 0);
      __builtin_amdgcn_s_wait_asynccnt(0);
    }
#else
    {
      const uint4* src =
          (const uint4*)(const void*)(hbuf + (size_t)(t & 1) * (B_ * H_));
      uint4* dst = (uint4*)(void*)h_lds;
#pragma unroll
      for (int q = 0; q < 4; ++q) dst[tid * 4 + q] = src[tid * 4 + q];
    }
#endif
    __syncthreads();

    // ---- 16x16 gate pre-activation tile: h(t) x U^T slice ----
    v8f acc = {};
    const __bf16* arow = h_lds + lrow * H_;
#pragma unroll
    for (int kt = 0; kt < 16; ++kt) {
      const int kb = kt * 32 + lhi * 8;
      v16bf a = frag_from_lds(arow + kb, arow + kb + 16);
      acc = __builtin_amdgcn_wmma_f32_16x16x32_bf16(
          false, a, false, ufrag[kt], (short)0, acc, false, false);
    }

    // ---- add wx(t) + u_bias; park in LDS for the gate combine ----
    const float* wxcol = wx + (size_t)t * G_ + nme;
#pragma unroll
    for (int r = 0; r < 8; ++r) {
      const int m = r + lhi * 8;  // batch index (WMMA C-layout)
      gates_lds[wv][m * 16 + lrow] =
          acc[r] + ubias + wxcol[(size_t)m * T_ * G_];
    }
    if (t + 1 < T_) __builtin_prefetch(wxcol + G_, 0, 1);  // global_prefetch_b8
    __syncthreads();

    // ---- elementwise cell update (2 elements per thread) ----
#pragma unroll
    for (int q = 0; q < 2; ++q) {
      const int col = cc + q;
      const int tl  = col >> 4;
      const int nl  = col & 15;
      const int idx = b_el * 16 + nl;
      const float gi = gates_lds[0 * 2 + tl][idx];
      const float gf = gates_lds[1 * 2 + tl][idx];
      const float gg = gates_lds[2 * 2 + tl][idx];
      const float go = gates_lds[3 * 2 + tl][idx];
      const float i_ = 1.f / (1.f + __expf(-gi));
      const float f_ = 1.f / (1.f + __expf(-gf));
      const float g_ = tanhf(gg);
      const float o_ = 1.f / (1.f + __expf(-go));
      const float c_ = f_ * c_reg[q] + i_ * g_;
      const float h_ = o_ * tanhf(c_);
      c_reg[q] = c_;
      const size_t oidx = ((size_t)b_el * T_ + t) * H_ + hc0 + col;
      out_h[oidx] = h_;
      out_c[oidx] = c_;
      hbuf[(size_t)((t + 1) & 1) * (B_ * H_) + b_el * H_ + hc0 + col] = (__bf16)h_;
    }

    // ---- device-wide barrier (monotonic counter, no reset races) ----
    __threadfence();
    __syncthreads();
    if (tid == 0) {
      __hip_atomic_fetch_add(bar_cnt, 1u, __ATOMIC_ACQ_REL,
                             __HIP_MEMORY_SCOPE_AGENT);
      const unsigned target = 16u * (unsigned)(t + 1);
      while (__hip_atomic_load(bar_cnt, __ATOMIC_ACQUIRE,
                               __HIP_MEMORY_SCOPE_AGENT) < target)
        __builtin_amdgcn_s_sleep(1);
    }
    __syncthreads();
  }
}

// ---------------------------------------------------------------------------
extern "C" void kernel_launch(void* const* d_in, const int* in_sizes, int n_in,
                              void* d_out, int out_size, void* d_ws,
                              size_t ws_size, hipStream_t stream) {
  (void)in_sizes; (void)n_in; (void)out_size; (void)ws_size;
  const float* x     = (const float*)d_in[0];  // [16,2048,512]
  const float* h0    = (const float*)d_in[1];  // [16,512]
  const float* c0    = (const float*)d_in[2];  // [16,512]
  const float* w     = (const float*)d_in[3];  // [2048,512]
  const float* wb    = (const float*)d_in[4];  // [2048]
  const float* u     = (const float*)d_in[5];  // [2048,512]
  const float* ubias = (const float*)d_in[6];  // [2048]
  float* out = (float*)d_out;

  unsigned char* ws = (unsigned char*)d_ws;
  unsigned* cnt = (unsigned*)ws;                     // barrier counter
  __bf16* hbuf  = (__bf16*)(ws + 512);               // 32 KB h double buffer
  float* wx     = (float*)(ws + (size_t)(1u << 16)); // 256 MiB wx scratch

  (void)hipMemsetAsync(ws, 0, 512, stream);                  // zero barrier
  init_kernel<<<1, 256, 0, stream>>>(h0, hbuf);              // h(0) -> bf16
  gemm_wx_kernel<<<2048, 256, 0, stream>>>(x, w, wb, wx);    // input proj
  lstm_scan_kernel<<<16, 256, 0, stream>>>(wx, u, ubias, c0, hbuf, cnt, out);
}
